// TFLSHAttention_85066122265699
// MI455X (gfx1250) — matrix operand order repaired and checked by reading, not compiled, not run on hardware
//
#include <hip/hip_runtime.h>
#include <hip/hip_bf16.h>
#include <math.h>

typedef _Float16 v16h __attribute__((ext_vector_type(16)));
typedef _Float16 v8h  __attribute__((ext_vector_type(8)));
typedef _Float16 h2   __attribute__((ext_vector_type(2)));
typedef float    v8f  __attribute__((ext_vector_type(8)));

#define B_ 8
#define S_ 8192
#define D_ 64
#define H_ 8
#define NBUCKETS 128
#define NCHUNKS (H_*NBUCKETS)   /* 1024 chunks of 64 per batch */

/* ---------------- Kernel 1: LSH hashing ---------------- */
/* buckets[b][h][s] = argmax over [r, -r] of qk[b,s,:] @ rot[:,h,:] */
__global__ __launch_bounds__(256) void lsh_hash_kernel(
    const float* __restrict__ qk, const float* __restrict__ rot,
    int* __restrict__ buckets) {
  __shared__ float rotLDS[64 * 64];                 /* [f][i], 16 KB */
  const int tid = threadIdx.x;
  const long gid = (long)blockIdx.x * 256 + tid;    /* b*S + s */
  const int b = (int)(gid >> 13);
  const int s = (int)(gid & 8191);

  float q[64];
  const float4* qr = (const float4*)(qk + (size_t)gid * 64);
#pragma unroll
  for (int i = 0; i < 16; ++i) {
    float4 t = qr[i];
    q[4*i+0] = t.x; q[4*i+1] = t.y; q[4*i+2] = t.z; q[4*i+3] = t.w;
  }

  for (int h = 0; h < H_; ++h) {
    __syncthreads();
    for (int idx = tid; idx < 4096; idx += 256) {   /* rot[f][h][i] -> LDS[f][i] */
      int f = idx >> 6, i = idx & 63;
      rotLDS[idx] = rot[f * (H_*64) + h * 64 + i];
    }
    __syncthreads();
    float bp = -3.0e38f, bn = -3.0e38f;
    int ip = 0, inn = 0;
    for (int i = 0; i < 64; ++i) {
      float acc = 0.f;
#pragma unroll 8
      for (int f = 0; f < 64; ++f) acc = fmaf(q[f], rotLDS[f * 64 + i], acc);
      if (acc > bp)  { bp = acc;  ip  = i; }
      if (-acc > bn) { bn = -acc; inn = i; }
    }
    /* positive half comes first in the concat -> wins ties */
    int bucket = (bp >= bn) ? ip : (64 + inn);
    buckets[((size_t)b * H_ + h) * S_ + s] = bucket;
  }
}

/* ---------------- Kernel 2: stable counting sort per (b,h) ---------------- */
/* st[b][h][dst] = original position s, sorted by (bucket, s) ascending */
__global__ __launch_bounds__(128) void lsh_sort_kernel(
    const int* __restrict__ buckets, int* __restrict__ st) {
  __shared__ int hist[128 * 128];                   /* [threadRow][bucket], 64 KB */
  __shared__ int base[128];
  const int t = threadIdx.x;
  const int* bkt = buckets + (size_t)blockIdx.x * S_;
  int* out = st + (size_t)blockIdx.x * S_;

  for (int i = 0; i < 128; ++i) hist[t * 128 + i] = 0;
  for (int j = 0; j < 64; ++j) hist[t * 128 + bkt[t * 64 + j]]++;
  __syncthreads();

  /* column-wise exclusive scan (thread t owns bucket t) */
  int run = 0;
  for (int row = 0; row < 128; ++row) {
    int c = hist[row * 128 + t];
    hist[row * 128 + t] = run;
    run += c;
  }
  base[t] = run;                                    /* bucket totals */
  __syncthreads();
  if (t == 0) {                                     /* exclusive scan of totals */
    int acc = 0;
    for (int i = 0; i < 128; ++i) { int c = base[i]; base[i] = acc; acc += c; }
  }
  __syncthreads();

  for (int j = 0; j < 64; ++j) {                    /* stable scatter */
    int s  = t * 64 + j;
    int bk = bkt[s];
    int ofs = hist[t * 128 + bk];
    hist[t * 128 + bk] = ofs + 1;
    out[base[bk] + ofs] = s;
  }
}

/* ---------------- Kernel 3: chunk attention with WMMA ---------------- */
__global__ __launch_bounds__(128) void lsh_attn_kernel(
    const float* __restrict__ qk, const float* __restrict__ v,
    const int* __restrict__ st, float* __restrict__ o_hash,
    float* __restrict__ logits) {
  __shared__ __attribute__((aligned(16))) _Float16 qh[64 * 72];    /* q*0.125, [m][k]  */
  __shared__ __attribute__((aligned(16))) _Float16 kh[128 * 72];   /* k-hat,   [kv][k] */
  __shared__ __attribute__((aligned(16))) _Float16 vT[64 * 136];   /* v^T,     [f][kv] */
  __shared__ __attribute__((aligned(16))) _Float16 pr[64 * 136];   /* probs,   [m][kv] */
  __shared__ int tkv[128];

  const int b = blockIdx.x >> 10;
  const int c = blockIdx.x & (NCHUNKS - 1);
  const int cprev = (c + NCHUNKS - 1) & (NCHUNKS - 1);  /* look-one-back, wraps */
  const int h = c >> 7;
  const int tid = threadIdx.x;
  const int* stB = st + (size_t)b * (H_ * S_);

  /* ---- gather phase: 128 threads, one kv row each ---- */
  {
    const int j = tid;
    const int g = ((j < 64) ? c : cprev) * 64 + (j & 63);
    const int p = stB[g];
    tkv[j] = p;

    const float4* qr = (const float4*)(qk + ((size_t)b * S_ + p) * 64);
    float row[64];
    float ss = 0.f;
#pragma unroll
    for (int i = 0; i < 16; ++i) {
      float4 tv = qr[i];
      row[4*i+0] = tv.x; row[4*i+1] = tv.y; row[4*i+2] = tv.z; row[4*i+3] = tv.w;
      ss += tv.x*tv.x + tv.y*tv.y + tv.z*tv.z + tv.w*tv.w;
    }
    const float rinv = rsqrtf(ss + 1e-6f);
#pragma unroll
    for (int k = 0; k < 64; k += 2) {
      h2 pk; pk.x = (_Float16)(row[k] * rinv); pk.y = (_Float16)(row[k+1] * rinv);
      *(h2*)&kh[j * 72 + k] = pk;
    }
    if (j < 64) {
#pragma unroll
      for (int k = 0; k < 64; k += 2) {
        h2 pq; pq.x = (_Float16)(row[k] * 0.125f); pq.y = (_Float16)(row[k+1] * 0.125f);
        *(h2*)&qh[j * 72 + k] = pq;
      }
    }
    const float4* vr = (const float4*)(v + ((size_t)b * S_ + p) * 64);
#pragma unroll
    for (int i = 0; i < 16; ++i) {
      float4 tv = vr[i];
      vT[(4*i+0) * 136 + j] = (_Float16)tv.x;
      vT[(4*i+1) * 136 + j] = (_Float16)tv.y;
      vT[(4*i+2) * 136 + j] = (_Float16)tv.z;
      vT[(4*i+3) * 136 + j] = (_Float16)tv.w;
    }
  }
  __syncthreads();

  const int lane = tid & 31;
  const int w    = tid >> 5;       /* wave -> 16-row strip of M */
  const int hlf  = lane >> 4;      /* A: K-group select; B: K-group select */
  const int lr   = lane & 15;      /* A: row; B/C/D: column */

  /* ---- GEMM1: dots = (q*0.125) @ khat^T  -> 64x128 ---- */
  v16h a[2];
#pragma unroll
  for (int kc = 0; kc < 2; ++kc) {
    const _Float16* ap = &qh[(w*16 + lr) * 72 + kc*32 + hlf*8];
    v8h lo = *(const v8h*)ap;
    v8h hi = *(const v8h*)(ap + 16);
    a[kc] = __builtin_shufflevector(lo, hi, 0,1,2,3,4,5,6,7,8,9,10,11,12,13,14,15);
  }

  v8f acc[8] = {};
#pragma unroll
  for (int nt = 0; nt < 8; ++nt) {
#pragma unroll
    for (int kc = 0; kc < 2; ++kc) {
      const _Float16* bp = &kh[(nt*16 + lr) * 72 + kc*32 + hlf*8];
      v8h lo = *(const v8h*)bp;
      v8h hi = *(const v8h*)(bp + 16);
      v16h bb = __builtin_shufflevector(lo, hi, 0,1,2,3,4,5,6,7,8,9,10,11,12,13,14,15);
      acc[nt] = __builtin_amdgcn_wmma_f32_16x16x32_f16(
          false, a[kc], false, bb, (short)0, acc[nt], false, false);
    }
  }

  /* ---- masks + row softmax (LSE) ---- */
  int rowt[8], colt[8];
#pragma unroll
  for (int r = 0; r < 8; ++r)  rowt[r]  = tkv[w*16 + hlf*8 + r];
#pragma unroll
  for (int nt = 0; nt < 8; ++nt) colt[nt] = tkv[nt*16 + lr];

  float lse[8];
#pragma unroll
  for (int r = 0; r < 8; ++r) {
    float vals[8];
    float mx = -3.0e38f;
#pragma unroll
    for (int nt = 0; nt < 8; ++nt) {
      float x = acc[nt][r];
      x = (colt[nt] >  rowt[r]) ? -1.0e38f   : x;   /* causal */
      x = (colt[nt] == rowt[r]) ? -50000.0f  : x;   /* self   */
      vals[nt] = x;
      mx = fmaxf(mx, x);
    }
    mx = fmaxf(mx, __shfl_xor(mx, 1, 32));
    mx = fmaxf(mx, __shfl_xor(mx, 2, 32));
    mx = fmaxf(mx, __shfl_xor(mx, 4, 32));
    mx = fmaxf(mx, __shfl_xor(mx, 8, 32));
    float s = 0.f;
#pragma unroll
    for (int nt = 0; nt < 8; ++nt) s += __expf(vals[nt] - mx);
    s += __shfl_xor(s, 1, 32); s += __shfl_xor(s, 2, 32);
    s += __shfl_xor(s, 4, 32); s += __shfl_xor(s, 8, 32);
    const float l = mx + __logf(s);
    lse[r] = l;
#pragma unroll
    for (int nt = 0; nt < 8; ++nt)
      pr[(w*16 + hlf*8 + r) * 136 + nt*16 + lr] = (_Float16)__expf(vals[nt] - l);
  }
  __syncthreads();

  /* ---- GEMM2: out = probs @ v  -> 64x64 ---- */
  v8f oacc[4] = {};
#pragma unroll
  for (int kc = 0; kc < 4; ++kc) {
    const _Float16* ap = &pr[(w*16 + lr) * 136 + kc*32 + hlf*8];
    v8h alo = *(const v8h*)ap;
    v8h ahi = *(const v8h*)(ap + 16);
    v16h a2 = __builtin_shufflevector(alo, ahi, 0,1,2,3,4,5,6,7,8,9,10,11,12,13,14,15);
#pragma unroll
    for (int nt = 0; nt < 4; ++nt) {
      const _Float16* bp = &vT[(nt*16 + lr) * 136 + kc*32 + hlf*8];
      v8h blo = *(const v8h*)bp;
      v8h bhi = *(const v8h*)(bp + 16);
      v16h bb = __builtin_shufflevector(blo, bhi, 0,1,2,3,4,5,6,7,8,9,10,11,12,13,14,15);
      oacc[nt] = __builtin_amdgcn_wmma_f32_16x16x32_f16(
          false, a2, false, bb, (short)0, oacc[nt], false, false);
    }
  }

  /* ---- scatter to original positions (fused unsort) ---- */
  const size_t baseBH = ((size_t)b * H_ + h) * S_;
#pragma unroll
  for (int r = 0; r < 8; ++r) {
    const size_t orow = (baseBH + (size_t)rowt[r]) * 64;
#pragma unroll
    for (int nt = 0; nt < 4; ++nt)
      o_hash[orow + nt*16 + lr] = oacc[nt][r];
  }
  if (lr == 0) {
#pragma unroll
    for (int r = 0; r < 8; ++r) logits[baseBH + rowt[r]] = lse[r];
  }
}

/* ---------------- Kernel 4: combine hash rounds ---------------- */
__global__ __launch_bounds__(64) void lsh_combine_kernel(
    const float* __restrict__ o_hash, const float* __restrict__ logits,
    float* __restrict__ out) {
  const int bs = blockIdx.x;        /* b*S + s */
  const int b = bs >> 13;
  const int s = bs & 8191;
  const int f = threadIdx.x;

  float lg[8];
#pragma unroll
  for (int h = 0; h < 8; ++h) lg[h] = logits[((size_t)b * H_ + h) * S_ + s];
  float mx = lg[0];
#pragma unroll
  for (int h = 1; h < 8; ++h) mx = fmaxf(mx, lg[h]);
  float e[8], sum = 0.f;
#pragma unroll
  for (int h = 0; h < 8; ++h) { e[h] = __expf(lg[h] - mx); sum += e[h]; }
  const float inv = 1.0f / sum;
  float acc = 0.f;
#pragma unroll
  for (int h = 0; h < 8; ++h)
    acc += e[h] * o_hash[(((size_t)b * H_ + h) * S_ + s) * 64 + f];
  out[((size_t)b * S_ + s) * 64 + f] = acc * inv;
}

extern "C" void kernel_launch(void* const* d_in, const int* in_sizes, int n_in,
                              void* d_out, int out_size, void* d_ws, size_t ws_size,
                              hipStream_t stream) {
  const float* qk  = (const float*)d_in[0];
  const float* v   = (const float*)d_in[1];
  const float* rot = (const float*)d_in[2];
  float* out = (float*)d_out;

  char* ws = (char*)d_ws;
  int*   buckets = (int*)  (ws);                            /* 2 MB   */
  int*   st      = (int*)  (ws + ((size_t)2 << 20));        /* 2 MB   */
  float* logits  = (float*)(ws + ((size_t)4 << 20));        /* 2 MB   */
  float* o_hash  = (float*)(ws + ((size_t)6 << 20));        /* 134 MB */

  lsh_hash_kernel   <<<(B_ * S_) / 256, 256, 0, stream>>>(qk, rot, buckets);
  lsh_sort_kernel   <<<B_ * H_,        128, 0, stream>>>(buckets, st);
  lsh_attn_kernel   <<<B_ * NCHUNKS,   128, 0, stream>>>(qk, v, st, o_hash, logits);
  lsh_combine_kernel<<<B_ * S_,         64, 0, stream>>>(o_hash, logits, out);
}
